// MC_Loss_9028021256444
// MI455X (gfx1250) — matrix operand order repaired and checked by reading, not compiled, not run on hardware
//
#include <hip/hip_runtime.h>
#include <hip/hip_bf16.h>

typedef _Float16 v16h __attribute__((ext_vector_type(16)));
typedef _Float16 v8h  __attribute__((ext_vector_type(8)));
typedef float    v8f  __attribute__((ext_vector_type(8)));

#define NBATCH 4
#define NP     2048      // rows per batch
#define DIM    256
#define ITERS  50
#define NCH    64        // row-chunks for the column pass (NP/NCH = 32 rows/chunk)
#define STAB   1e-8f
#define MU     (1.0f / 2048.0f)

__device__ __forceinline__ float wave_sum(float x) {
    #pragma unroll
    for (int off = 16; off > 0; off >>= 1) x += __shfl_xor(x, off, 32);
    return x;
}

// ---------------------------------------------------------------------------
// 1) Row-normalize f32 [rows,256] -> f16 [rows,256].  One wave per row.
// ---------------------------------------------------------------------------
__global__ __launch_bounds__(256) void normalize_kernel(const float* __restrict__ x,
                                                        _Float16* __restrict__ out,
                                                        int rows) {
    int wave = threadIdx.x >> 5;
    int lane = threadIdx.x & 31;
    int row  = blockIdx.x * 8 + wave;
    if (row >= rows) return;
    const float* xp = x + (size_t)row * DIM + lane * 8;
    v8f v = *(const v8f*)xp;
    float ss = 0.f;
    #pragma unroll
    for (int e = 0; e < 8; ++e) ss += v[e] * v[e];
    ss = wave_sum(ss);
    float scale = 1.0f / (sqrtf(ss) + STAB);
    v8h o;
    #pragma unroll
    for (int e = 0; e < 8; ++e) o[e] = (_Float16)(v[e] * scale);
    *(v8h*)(out + (size_t)row * DIM + lane * 8) = o;
}

// ---------------------------------------------------------------------------
// 2) Per-batch S = Xn * Yn^T (2048x2048x256), K = exp(S-1), stored f16.
//    Block = 256 threads = 8 waves; block tile 128(M) x 64(N);
//    each wave: 16(M) x 64(N), software-pipelined fragment loads.
//    grid = (2048/64, 2048/128, NBATCH)
// ---------------------------------------------------------------------------
__global__ __launch_bounds__(256) void gemm_exp_kernel(const _Float16* __restrict__ X,
                                                       const _Float16* __restrict__ Y,
                                                       _Float16* __restrict__ Kout) {
    int wave = threadIdx.x >> 5;
    int lane = threadIdx.x & 31;
    int b    = blockIdx.z;
    int m_base = blockIdx.y * 128 + wave * 16;
    int n_base = blockIdx.x * 64;

    const _Float16* Xb = X + (size_t)b * NP * DIM;
    const _Float16* Yb = Y + (size_t)b * NP * DIM;

    int r   = lane & 15;         // A-fragment row within tile
    int sel = lane >> 4;         // lane-group selector (ISA 16-bit layouts)

    const _Float16* arow = Xb + (size_t)(m_base + r) * DIM;
    const _Float16* brow0 = Yb + (size_t)(n_base +  0 + r) * DIM;
    const _Float16* brow1 = Yb + (size_t)(n_base + 16 + r) * DIM;
    const _Float16* brow2 = Yb + (size_t)(n_base + 32 + r) * DIM;
    const _Float16* brow3 = Yb + (size_t)(n_base + 48 + r) * DIM;

    v8f acc0 = {}, acc1 = {}, acc2 = {}, acc3 = {};

    // prologue: fragments for k-step 0
    v8h  alo = *(const v8h*)(arow + sel * 16);
    v8h  ahi = *(const v8h*)(arow + sel * 16 + 8);
    v16h b0  = *(const v16h*)(brow0 + sel * 16);
    v16h b1  = *(const v16h*)(brow1 + sel * 16);
    v16h b2  = *(const v16h*)(brow2 + sel * 16);
    v16h b3  = *(const v16h*)(brow3 + sel * 16);

    #pragma unroll
    for (int kk = 0; kk < 8; ++kk) {
        v16h af;
        #pragma unroll
        for (int e = 0; e < 8; ++e) { af[e] = alo[e]; af[e + 8] = ahi[e]; }

        // prefetch next k-step before issuing this step's WMMAs
        v8h nalo = alo, nahi = ahi;
        v16h nb0 = b0, nb1 = b1, nb2 = b2, nb3 = b3;
        if (kk < 7) {
            int kb = (kk + 1) * 32;
            nalo = *(const v8h*)(arow + kb + sel * 16);
            nahi = *(const v8h*)(arow + kb + sel * 16 + 8);
            int ko = kb + sel * 16;
            nb0 = *(const v16h*)(brow0 + ko);
            nb1 = *(const v16h*)(brow1 + ko);
            nb2 = *(const v16h*)(brow2 + ko);
            nb3 = *(const v16h*)(brow3 + ko);
        }

        acc0 = __builtin_amdgcn_wmma_f32_16x16x32_f16(false, af, false, b0, (short)0, acc0, false, false);
        acc1 = __builtin_amdgcn_wmma_f32_16x16x32_f16(false, af, false, b1, (short)0, acc1, false, false);
        acc2 = __builtin_amdgcn_wmma_f32_16x16x32_f16(false, af, false, b2, (short)0, acc2, false, false);
        acc3 = __builtin_amdgcn_wmma_f32_16x16x32_f16(false, af, false, b3, (short)0, acc3, false, false);

        alo = nalo; ahi = nahi; b0 = nb0; b1 = nb1; b2 = nb2; b3 = nb3;
    }

    // C/D layout: VGPR i -> M = i + 8*sel ; N = n_base + nsub*16 + (lane&15)
    _Float16* Kb = Kout + (size_t)b * NP * NP;
    #pragma unroll
    for (int i = 0; i < 8; ++i) {
        int row = m_base + i + 8 * sel;
        size_t base = (size_t)row * NP + (size_t)(n_base + (lane & 15));
        Kb[base +  0] = (_Float16)__expf(acc0[i] - 1.0f);
        Kb[base + 16] = (_Float16)__expf(acc1[i] - 1.0f);
        Kb[base + 32] = (_Float16)__expf(acc2[i] - 1.0f);
        Kb[base + 48] = (_Float16)__expf(acc3[i] - 1.0f);
    }
}

// ---------------------------------------------------------------------------
// 3a) u_i = mu / (sum_j K[i][j] v[j] + STAB).  One wave per row.
//     grid = (NP/8, NBATCH), block 256.
// ---------------------------------------------------------------------------
__global__ __launch_bounds__(256) void sinkhorn_row_kernel(const _Float16* __restrict__ K,
                                                           const float* __restrict__ vin,
                                                           float* __restrict__ uout) {
    int wave = threadIdx.x >> 5;
    int lane = threadIdx.x & 31;
    int b    = blockIdx.y;
    int row  = blockIdx.x * 8 + wave;
    const _Float16* Kp = K + ((size_t)b * NP + row) * NP;
    const float*    vp = vin + (size_t)b * NP;
    float sum = 0.f;
    #pragma unroll
    for (int k = 0; k < 8; ++k) {
        int idx = (k * 32 + lane) * 8;
        v8h kv = *(const v8h*)(Kp + idx);
        v8f vv = *(const v8f*)(vp + idx);
        #pragma unroll
        for (int e = 0; e < 8; ++e) sum += (float)kv[e] * vv[e];
    }
    sum = wave_sum(sum);
    if (lane == 0) uout[(size_t)b * NP + row] = MU / (sum + STAB);
}

// ---------------------------------------------------------------------------
// 3b-1) Column-pass partials: vpart[b][chunk][j] = sum_{i in chunk} K[i][j]*u[i]
//       grid = (NCH, NBATCH), block 256; thread owns 8 consecutive columns.
//       Deterministic (no atomics); every element written each iteration.
// ---------------------------------------------------------------------------
__global__ __launch_bounds__(256) void sinkhorn_col_partial_kernel(const _Float16* __restrict__ K,
                                                                   const float* __restrict__ uin,
                                                                   float* __restrict__ vpart) {
    int b  = blockIdx.y;
    int ch = blockIdx.x;
    int c0 = threadIdx.x * 8;
    const _Float16* Kb = K + (size_t)b * NP * NP;
    const float*    up = uin + (size_t)b * NP;
    float s[8] = {};
    int i0 = ch * (NP / NCH);
    for (int i = i0; i < i0 + NP / NCH; ++i) {
        v8h kv = *(const v8h*)(Kb + (size_t)i * NP + c0);
        float ui = up[i];
        #pragma unroll
        for (int e = 0; e < 8; ++e) s[e] += ui * (float)kv[e];
    }
    float* pp = vpart + ((size_t)b * NCH + ch) * NP + c0;
    #pragma unroll
    for (int e = 0; e < 8; ++e) pp[e] = s[e];
}

// ---------------------------------------------------------------------------
// 3b-2) Finalize: v[b][j] = nu / (sum_ch vpart[b][ch][j] + STAB)
//       grid = NBATCH*NP/256 blocks.
// ---------------------------------------------------------------------------
__global__ __launch_bounds__(256) void sinkhorn_col_finalize_kernel(const float* __restrict__ vpart,
                                                                    float* __restrict__ vout) {
    int t = blockIdx.x * blockDim.x + threadIdx.x;   // t = b*NP + j
    int b = t >> 11;
    int j = t & (NP - 1);
    const float* pp = vpart + (size_t)b * NCH * NP + j;
    float s = 0.f;
    #pragma unroll 8
    for (int ch = 0; ch < NCH; ++ch) s += pp[(size_t)ch * NP];
    vout[t] = MU / (s + STAB);
}

// ---------------------------------------------------------------------------
// 4) Init v1=v2=1.0
// ---------------------------------------------------------------------------
__global__ __launch_bounds__(256) void init_v_kernel(float* v1, float* v2) {
    int t = blockIdx.x * blockDim.x + threadIdx.x;
    if (t < NBATCH * NP) { v1[t] = 1.0f; v2[t] = 1.0f; }
}

// ---------------------------------------------------------------------------
// 5) loss partial: mean |u1_i K1_ij v1_j - u2_i K2_ij v2_j|
// ---------------------------------------------------------------------------
__global__ __launch_bounds__(256) void reduce_l1_kernel(const _Float16* __restrict__ K1,
                                                        const _Float16* __restrict__ K2,
                                                        const float* __restrict__ u1,
                                                        const float* __restrict__ v1,
                                                        const float* __restrict__ u2,
                                                        const float* __restrict__ v2,
                                                        float* __restrict__ out) {
    const size_t nchunks = (size_t)NBATCH * NP * NP / 8;   // 8 elems per chunk
    const float scale = 1.0f / ((float)NBATCH * NP * NP);
    size_t stride = (size_t)gridDim.x * blockDim.x;
    float acc = 0.f;
    for (size_t ch = blockIdx.x * blockDim.x + threadIdx.x; ch < nchunks; ch += stride) {
        size_t rflat = ch >> 8;           // NP/8 = 256 chunks per row
        int    j     = (int)(ch & 255) * 8;
        int    b     = (int)(rflat >> 11);
        int    i     = (int)(rflat & (NP - 1));
        size_t base  = rflat * NP + j;
        v8h k1 = *(const v8h*)(K1 + base);
        v8h k2 = *(const v8h*)(K2 + base);
        float a1 = u1[(size_t)b * NP + i];
        float a2 = u2[(size_t)b * NP + i];
        const float* w1 = v1 + (size_t)b * NP + j;
        const float* w2 = v2 + (size_t)b * NP + j;
        #pragma unroll
        for (int e = 0; e < 8; ++e)
            acc += fabsf(a1 * (float)k1[e] * w1[e] - a2 * (float)k2[e] * w2[e]);
    }
    acc = wave_sum(acc);
    __shared__ float smem[8];
    int wave = threadIdx.x >> 5, lane = threadIdx.x & 31;
    if (lane == 0) smem[wave] = acc;
    __syncthreads();
    if (wave == 0) {
        float t = (lane < 8) ? smem[lane] : 0.f;
        t = wave_sum(t);
        if (lane == 0) atomicAdd(out, t * scale);
    }
}

// ---------------------------------------------------------------------------
extern "C" void kernel_launch(void* const* d_in, const int* in_sizes, int n_in,
                              void* d_out, int out_size, void* d_ws, size_t ws_size,
                              hipStream_t stream) {
    const float* src = (const float*)d_in[0];
    const float* tgt = (const float*)d_in[1];
    const float* gen = (const float*)d_in[2];
    float* out = (float*)d_out;

    char* ws = (char*)d_ws;
    const size_t KBYTES = (size_t)NBATCH * NP * NP * sizeof(_Float16);   // 32 MB
    const size_t FBYTES = (size_t)NBATCH * NP * DIM * sizeof(_Float16);  // 4 MB
    const size_t VBYTES = (size_t)NBATCH * NP * sizeof(float);           // 32 KB
    const size_t PBYTES = (size_t)NBATCH * NCH * NP * sizeof(float);     // 2 MB
    _Float16* K1 = (_Float16*)(ws);
    _Float16* K2 = (_Float16*)(ws + KBYTES);
    _Float16* Xs = (_Float16*)(ws + 2 * KBYTES);
    _Float16* Xt = (_Float16*)(ws + 2 * KBYTES + FBYTES);
    _Float16* Xg = (_Float16*)(ws + 2 * KBYTES + 2 * FBYTES);
    float* u1 = (float*)(ws + 2 * KBYTES + 3 * FBYTES);
    float* v1 = (float*)(ws + 2 * KBYTES + 3 * FBYTES + VBYTES);
    float* u2 = (float*)(ws + 2 * KBYTES + 3 * FBYTES + 2 * VBYTES);
    float* v2 = (float*)(ws + 2 * KBYTES + 3 * FBYTES + 3 * VBYTES);
    float* vpart = (float*)(ws + 2 * KBYTES + 3 * FBYTES + 4 * VBYTES);

    hipMemsetAsync(out, 0, sizeof(float), stream);

    // 1) normalize to f16
    int rows = NBATCH * NP;  // 8192
    normalize_kernel<<<rows / 8, 256, 0, stream>>>(src, Xs, rows);
    normalize_kernel<<<rows / 8, 256, 0, stream>>>(tgt, Xt, rows);
    normalize_kernel<<<rows / 8, 256, 0, stream>>>(gen, Xg, rows);

    // 2) Gibbs kernels via WMMA GEMM + exp epilogue
    dim3 ggrid(NP / 64, NP / 128, NBATCH);
    gemm_exp_kernel<<<ggrid, 256, 0, stream>>>(Xs, Xt, K1);
    gemm_exp_kernel<<<ggrid, 256, 0, stream>>>(Xt, Xg, K2);

    // 3) Sinkhorn
    init_v_kernel<<<(NBATCH * NP + 255) / 256, 256, 0, stream>>>(v1, v2);
    dim3 rgrid(NP / 8, NBATCH);
    dim3 cgrid(NCH, NBATCH);
    int  fgrid = NBATCH * NP / 256;
    for (int it = 0; it < ITERS; ++it) {
        sinkhorn_row_kernel<<<rgrid, 256, 0, stream>>>(K1, v1, u1);
        sinkhorn_col_partial_kernel<<<cgrid, 256, 0, stream>>>(K1, u1, vpart);
        sinkhorn_col_finalize_kernel<<<fgrid, 256, 0, stream>>>(vpart, v1);
        sinkhorn_row_kernel<<<rgrid, 256, 0, stream>>>(K2, v2, u2);
        sinkhorn_col_partial_kernel<<<cgrid, 256, 0, stream>>>(K2, u2, vpart);
        sinkhorn_col_finalize_kernel<<<fgrid, 256, 0, stream>>>(vpart, v2);
    }

    // 4) L1 mean of plan difference
    reduce_l1_kernel<<<2048, 256, 0, stream>>>(K1, K2, u1, v1, u2, v2, out);
}
